// IPANetModel_85383949845224
// MI455X (gfx1250) — matrix operand order repaired and checked by reading, not compiled, not run on hardware
//
#include <hip/hip_runtime.h>
#include <math.h>

typedef __attribute__((ext_vector_type(16))) __bf16 bf16x16;
typedef __attribute__((ext_vector_type(8)))  __bf16 bf16x8;
typedef __attribute__((ext_vector_type(4)))  __bf16 bf16x4;
typedef __attribute__((ext_vector_type(2)))  __bf16 bf16x2;
typedef __attribute__((ext_vector_type(8)))  float  f32x8;

#define NSEQ 256
#define CDIM 128
#define INF_ 100000.0f

// ---------------------------------------------------------------------------
// Tiled GEMM via v_wmma_f32_16x16x32_bf16.
// Block tile 128(M) x 64(N), 8 wave32s, each wave owns 32x32 (2x2 WMMA tiles).
// K stepped by 32, double-buffered LDS (40-element row pitch: 16B aligned,
// bank-conflict spread).  B is always bf16; weights are pre-converted AND
// pre-transposed once per launch so the hot path uses vectorized staging.
// Requires M % 128 == 0 (all call sites: 256 or 65536).
//
// AMODE: 0 = A is f32  [M][lda]
//        1 = A is bf16 [M][lda]
//        3 = A row r=(i*256+j) is bf16 concat[zb[i,j,0:128]|ne[i,0:64]|ne[j,0:64]]
// BMODE: 0 = B is bf16 [K][ldb]            (activation operand, transposed in LDS)
//        1 = B is bf16 [N][ldb]            (C = A * B^T; weights stored this way)
// ---------------------------------------------------------------------------
template <int AMODE, int BMODE>
__global__ __launch_bounds__(256) void gemm_kernel(
    const void* __restrict__ Ap, int lda, const __bf16* __restrict__ ne,
    const __bf16* __restrict__ Bb, int ldb,
    const float* __restrict__ bias,
    float* __restrict__ Cf, __bf16* __restrict__ Cb, int ldc,
    const float* __restrict__ resid, int ldr,
    int M, int N, int K, float alpha, int act)
{
    __shared__ __bf16 sA[2][128 * 40];
    __shared__ __bf16 sB[2][64 * 40];

    const int tid  = threadIdx.x;
    const int lane = tid & 31;
    const int wave = tid >> 5;
    const int wm   = wave >> 1;   // 0..3 -> M direction
    const int wn   = wave & 1;    // 0..1 -> N direction
    const int m0   = blockIdx.y * 128;
    const int n0   = blockIdx.x * 64;
    const int half = lane >> 4;
    const int l15  = lane & 15;

    // ---- staging lambdas: load to registers first, then convert/store ----
    auto stageA = [&](int k0, __bf16* dst) {
        const int r  = tid >> 3;          // 0..31
        const int c4 = (tid & 7) * 4;     // 0..28
        if (AMODE == 0) {
            const float* A = (const float*)Ap;
            float4 v[4];
            #pragma unroll
            for (int rr = 0; rr < 4; ++rr)
                v[rr] = *(const float4*)(A + (size_t)(m0 + rr * 32 + r) * lda + k0 + c4);
            #pragma unroll
            for (int rr = 0; rr < 4; ++rr) {
                bf16x4 t;
                t[0] = (__bf16)v[rr].x; t[1] = (__bf16)v[rr].y;
                t[2] = (__bf16)v[rr].z; t[3] = (__bf16)v[rr].w;
                *(bf16x4*)(dst + (rr * 32 + r) * 40 + c4) = t;
            }
        } else if (AMODE == 1) {
            const __bf16* A = (const __bf16*)Ap;
            bf16x4 v[4];
            #pragma unroll
            for (int rr = 0; rr < 4; ++rr)
                v[rr] = *(const bf16x4*)(A + (size_t)(m0 + rr * 32 + r) * lda + k0 + c4);
            #pragma unroll
            for (int rr = 0; rr < 4; ++rr)
                *(bf16x4*)(dst + (rr * 32 + r) * 40 + c4) = v[rr];
        } else {  // AMODE 3: bf16 edge concat; branch is uniform in k0
            const __bf16* Zb = (const __bf16*)Ap;
            bf16x4 v[4];
            #pragma unroll
            for (int rr = 0; rr < 4; ++rr) {
                const int gm = m0 + rr * 32 + r;
                const int i = gm >> 8, j = gm & 255;
                const __bf16* src;
                if (k0 < 128)      src = Zb + (size_t)(i * 256 + j) * 128 + k0 + c4;
                else if (k0 < 192) src = ne + i * 64 + (k0 - 128) + c4;
                else               src = ne + j * 64 + (k0 - 192) + c4;
                v[rr] = *(const bf16x4*)src;
            }
            #pragma unroll
            for (int rr = 0; rr < 4; ++rr)
                *(bf16x4*)(dst + (rr * 32 + r) * 40 + c4) = v[rr];
        }
    };
    auto stageB = [&](int k0, __bf16* dst) {
        if (BMODE == 0) {
            const int n2 = (tid & 31) * 2;   // 0..62
            const int kk = tid >> 5;         // 0..7
            const bool nok = (n0 + n2) < N;  // N always even -> pair in/out together
            bf16x2 v[4];
            #pragma unroll
            for (int pass = 0; pass < 4; ++pass) {
                const int k = pass * 8 + kk;
                bf16x2 t; t[0] = (__bf16)0.f; t[1] = (__bf16)0.f;
                if (nok) t = *(const bf16x2*)(Bb + (size_t)(k0 + k) * ldb + n0 + n2);
                v[pass] = t;
            }
            #pragma unroll
            for (int pass = 0; pass < 4; ++pass) {
                const int k = pass * 8 + kk;
                dst[n2 * 40 + k]       = v[pass][0];
                dst[(n2 + 1) * 40 + k] = v[pass][1];
            }
        } else {  // B^T: bf16 [N][ldb] -> direct vectorized copy
            const int r  = tid >> 3;
            const int c4 = (tid & 7) * 4;
            bf16x4 v[2];
            #pragma unroll
            for (int rr = 0; rr < 2; ++rr) {
                const int n = rr * 32 + r;
                bf16x4 t; t[0] = (__bf16)0.f; t[1] = (__bf16)0.f; t[2] = (__bf16)0.f; t[3] = (__bf16)0.f;
                if (n0 + n < N) t = *(const bf16x4*)(Bb + (size_t)(n0 + n) * ldb + k0 + c4);
                v[rr] = t;
            }
            #pragma unroll
            for (int rr = 0; rr < 2; ++rr)
                *(bf16x4*)(dst + (rr * 32 + r) * 40 + c4) = v[rr];
        }
    };

    f32x8 acc[2][2] = {};

    stageA(0, sA[0]);
    stageB(0, sB[0]);
    int cur = 0;

    for (int k0 = 0; k0 < K; k0 += 32) {
        __syncthreads();  // staged tile visible; previous compute done
        if (k0 + 32 < K) {
            stageA(k0 + 32, sA[cur ^ 1]);
            stageB(k0 + 32, sB[cur ^ 1]);
        }
        // ---- fragments in WMMA lane layout ----
        bf16x16 afrag[2], bfrag[2];
        #pragma unroll
        for (int mi = 0; mi < 2; ++mi) {
            const int m = wm * 32 + mi * 16 + l15;
            const __bf16* base = &sA[cur][m * 40 + half * 8];
            bf16x8 lo = *(const bf16x8*)(base);
            bf16x8 hi = *(const bf16x8*)(base + 16);
            #pragma unroll
            for (int t = 0; t < 8; ++t) { afrag[mi][t] = lo[t]; afrag[mi][8 + t] = hi[t]; }
        }
        #pragma unroll
        for (int ni = 0; ni < 2; ++ni) {
            const int n = wn * 32 + ni * 16 + l15;
            const __bf16* base = &sB[cur][n * 40 + half * 8];
            bf16x8 lo = *(const bf16x8*)(base);
            bf16x8 hi = *(const bf16x8*)(base + 16);
            #pragma unroll
            for (int t = 0; t < 8; ++t) { bfrag[ni][t] = lo[t]; bfrag[ni][8 + t] = hi[t]; }
        }
        #pragma unroll
        for (int mi = 0; mi < 2; ++mi)
            #pragma unroll
            for (int ni = 0; ni < 2; ++ni)
                acc[mi][ni] = __builtin_amdgcn_wmma_f32_16x16x32_bf16(
                    false, afrag[mi], false, bfrag[ni], (short)0, acc[mi][ni], false, false);
        cur ^= 1;
    }

    // ---- epilogue: alpha*acc + bias -> relu -> + resid -> store ----
    #pragma unroll
    for (int mi = 0; mi < 2; ++mi) {
        #pragma unroll
        for (int ni = 0; ni < 2; ++ni) {
            const int col = n0 + wn * 32 + ni * 16 + l15;
            if (col >= N) continue;
            const float bv = bias ? bias[col] : 0.f;
            #pragma unroll
            for (int r = 0; r < 8; ++r) {
                const int row = m0 + wm * 32 + mi * 16 + half * 8 + r;
                float v = alpha * acc[mi][ni][r] + bv;
                if (act) v = fmaxf(v, 0.f);
                if (resid) v += resid[(size_t)row * ldr + col];
                if (Cf) Cf[(size_t)row * ldc + col] = v;
                else    Cb[(size_t)row * ldc + col] = (__bf16)v;
            }
        }
    }
}

// ---------------------------------------------------------------------------
// f32 -> bf16 bulk convert (z shadow)
// ---------------------------------------------------------------------------
__global__ __launch_bounds__(256) void cvt_kernel(
    const float* __restrict__ s, __bf16* __restrict__ d, int n)
{
    const int i = blockIdx.x * 256 + threadIdx.x;
    if (i < n) d[i] = (__bf16)s[i];
}

// ---------------------------------------------------------------------------
// f32 [K][N] -> bf16 [N][K] transpose-convert (weights, once per launch)
// ---------------------------------------------------------------------------
__global__ __launch_bounds__(256) void cvtT_kernel(
    const float* __restrict__ s, __bf16* __restrict__ d, int K, int N)
{
    const int i = blockIdx.x * 256 + threadIdx.x;  // output-major index: n*K + k
    if (i < K * N) {
        const int n = i / K, k = i - n * K;
        d[i] = (__bf16)s[(size_t)k * N + n];
    }
}

// ---------------------------------------------------------------------------
// LayerNorm over last dim (128 or 512); optional bf16 shadow output.
// ---------------------------------------------------------------------------
__global__ __launch_bounds__(128) void ln_kernel(
    const float* __restrict__ x, float* __restrict__ y, __bf16* __restrict__ yb,
    const float* __restrict__ g, const float* __restrict__ b, int dim)
{
    const int row = blockIdx.x, t = threadIdx.x;
    const int nv = dim >> 7;
    const float* xr = x + (size_t)row * dim;
    float v[4];
    float s = 0.f, ss = 0.f;
    for (int i = 0; i < nv; ++i) {
        float val = xr[t + i * 128];
        v[i] = val; s += val; ss += val * val;
    }
    __shared__ float r1[128], r2[128];
    r1[t] = s; r2[t] = ss; __syncthreads();
    for (int off = 64; off; off >>= 1) {
        if (t < off) { r1[t] += r1[t + off]; r2[t] += r2[t + off]; }
        __syncthreads();
    }
    const float mean = r1[0] / dim;
    const float var  = r2[0] / dim - mean * mean;
    const float inv  = rsqrtf(var + 1e-5f);
    float* yr = y + (size_t)row * dim;
    for (int i = 0; i < nv; ++i) {
        const int c = t + i * 128;
        const float o = (v[i] - mean) * inv * g[c] + b[c];
        yr[c] = o;
        if (yb) yb[(size_t)row * dim + c] = (__bf16)o;
    }
}

// ---------------------------------------------------------------------------
// Rotate/translate raw point projections; kv points also emitted in bf16.
// ---------------------------------------------------------------------------
__global__ __launch_bounds__(64) void pts_kernel(
    const float* __restrict__ qpraw, const float* __restrict__ kvpraw,
    const float* __restrict__ rot, const float* __restrict__ trans,
    float* __restrict__ qpts, float* __restrict__ kvpts, __bf16* __restrict__ kvptsb)
{
    const int n = blockIdx.x, t = threadIdx.x;
    float R[9], T[3];
    #pragma unroll
    for (int q = 0; q < 9; ++q) R[q] = rot[n * 9 + q];
    #pragma unroll
    for (int q = 0; q < 3; ++q) T[q] = trans[n * 3 + q];
    if (t < 16) {
        const float px = qpraw[n * 48 + t];
        const float py = qpraw[n * 48 + 16 + t];
        const float pz = qpraw[n * 48 + 32 + t];
        #pragma unroll
        for (int ii = 0; ii < 3; ++ii)
            qpts[n * 48 + t * 3 + ii] = R[ii * 3 + 0] * px + R[ii * 3 + 1] * py + R[ii * 3 + 2] * pz + T[ii];
    } else {
        const int p = t - 16;
        const float px = kvpraw[n * 144 + p];
        const float py = kvpraw[n * 144 + 48 + p];
        const float pz = kvpraw[n * 144 + 96 + p];
        #pragma unroll
        for (int ii = 0; ii < 3; ++ii) {
            const float o = R[ii * 3 + 0] * px + R[ii * 3 + 1] * py + R[ii * 3 + 2] * pz + T[ii];
            kvpts[n * 144 + p * 3 + ii]  = o;
            kvptsb[n * 144 + p * 3 + ii] = (__bf16)o;
        }
    }
}

// ---------------------------------------------------------------------------
// Fused attention bias + point attention + mask + softmax over j.
// ---------------------------------------------------------------------------
__global__ __launch_bounds__(256) void softmax_kernel(
    float* __restrict__ attn, const float* __restrict__ biasz,
    const float* __restrict__ qpts, const float* __restrict__ kvpts,
    const float* __restrict__ head_w, const float* __restrict__ mask)
{
    const int i = blockIdx.x, h = blockIdx.y, j = threadIdx.x;
    __shared__ float qp[12];
    __shared__ float red[256];
    if (j < 12) qp[j] = qpts[i * 48 + h * 12 + j];
    __syncthreads();

    const float x  = head_w[h];
    const float sp = (x > 20.f) ? x : log1pf(expf(x));
    const float hw = sp * 0.13608276348795434f;  // sqrt(1/54)

    float d2 = 0.f;
    #pragma unroll
    for (int pp = 0; pp < 4; ++pp) {
        const float* kp = &kvpts[j * 144 + (h * 12 + pp) * 3];
        const float dx = qp[pp * 3 + 0] - kp[0];
        const float dy = qp[pp * 3 + 1] - kp[1];
        const float dz = qp[pp * 3 + 2] - kp[2];
        d2 += dx * dx + dy * dy + dz * dz;
    }
    float* arow = attn + ((size_t)h * 65536 + (size_t)i * 256);
    float l = arow[j] + 0.5773502691896258f * biasz[(i * 256 + j) * 4 + h]
              - 0.5f * hw * d2 + INF_ * (mask[i] * mask[j] - 1.0f);

    red[j] = l; __syncthreads();
    for (int off = 128; off; off >>= 1) { if (j < off) red[j] = fmaxf(red[j], red[j + off]); __syncthreads(); }
    const float m = red[0]; __syncthreads();
    const float e = expf(l - m);
    red[j] = e; __syncthreads();
    for (int off = 128; off; off >>= 1) { if (j < off) red[j] += red[j + off]; __syncthreads(); }
    arow[j] = e / red[0];
}

// ---------------------------------------------------------------------------
// o_pair[i, h*128+c] = sum_j attn[h,i,j] * z[i,j,c]
// ---------------------------------------------------------------------------
__global__ __launch_bounds__(128) void opair_kernel(
    const float* __restrict__ attn, const float* __restrict__ z, float* __restrict__ feats)
{
    const int i = blockIdx.x, c = threadIdx.x;
    float a0 = 0.f, a1 = 0.f, a2 = 0.f, a3 = 0.f;
    const float* zrow = z + (size_t)i * 256 * 128;
    const float* ar   = attn + (size_t)i * 256;
    for (int j = 0; j < 256; ++j) {
        const float zv = zrow[j * 128 + c];
        a0 += ar[j] * zv;
        a1 += ar[65536 + j] * zv;
        a2 += ar[131072 + j] * zv;
        a3 += ar[196608 + j] * zv;
    }
    float* f = feats + (size_t)i * 768 + 256;
    f[c] = a0; f[128 + c] = a1; f[256 + c] = a2; f[384 + c] = a3;
}

// ---------------------------------------------------------------------------
// o_pt finalize: rot^T @ (o_pt - trans), norm; scatter into feats[128:256].
// ---------------------------------------------------------------------------
__global__ __launch_bounds__(32) void optfin_kernel(
    const float* __restrict__ optraw, const float* __restrict__ rot,
    const float* __restrict__ trans, float* __restrict__ feats)
{
    const int n = blockIdx.x, t = threadIdx.x;
    const int h = t >> 3, pp = t & 7;
    const float* o = optraw + n * 96 + h * 24 + pp * 3;
    const float vx = o[0] - trans[n * 3 + 0];
    const float vy = o[1] - trans[n * 3 + 1];
    const float vz = o[2] - trans[n * 3 + 2];
    const float* R = rot + n * 9;
    const float rx = R[0] * vx + R[3] * vy + R[6] * vz;
    const float ry = R[1] * vx + R[4] * vy + R[7] * vz;
    const float rz = R[2] * vx + R[5] * vy + R[8] * vz;
    const float nm = sqrtf(rx * rx + ry * ry + rz * rz + 1e-8f);
    float* f = feats + (size_t)n * 768;
    f[128 + t] = rx; f[160 + t] = ry; f[192 + t] = rz; f[224 + t] = nm;
}

// ---------------------------------------------------------------------------
// Host-side orchestration
// ---------------------------------------------------------------------------
struct LayerP {
    const float *b_b, *b_w, *e1_b, *e1_w, *e2_b, *e2_w, *e_f_b, *e_f_w;
    const float *e_init_b, *e_init_w, *e_ln_b, *e_ln_g;
    const float *head_w, *kv_b, *kv_w, *kvp_b, *kvp_w, *ln_b, *ln_g;
    const float *out_b, *out_w, *post_b, *post_w, *pre_b, *pre_w;
    const float *q_b, *q_w, *qp_b, *qp_w;
    const float *t1_b, *t1_w, *t2_b, *t2_w, *t3_b, *t3_w, *tln_b, *tln_g;
    bool edge;
};

static LayerP get_layer(void* const* din, int li) {
    LayerP p{};
    auto F = [&](int k) { return (const float*)din[k]; };
    if (li < 7) {  // jax pytree: dict keys sorted
        const int b = 6 + li * 37;
        p.b_b = F(b + 0);  p.b_w = F(b + 1);  p.e1_b = F(b + 2);  p.e1_w = F(b + 3);
        p.e2_b = F(b + 4); p.e2_w = F(b + 5); p.e_f_b = F(b + 6); p.e_f_w = F(b + 7);
        p.e_init_b = F(b + 8); p.e_init_w = F(b + 9); p.e_ln_b = F(b + 10); p.e_ln_g = F(b + 11);
        p.head_w = F(b + 12); p.kv_b = F(b + 13); p.kv_w = F(b + 14);
        p.kvp_b = F(b + 15); p.kvp_w = F(b + 16); p.ln_b = F(b + 17); p.ln_g = F(b + 18);
        p.out_b = F(b + 19); p.out_w = F(b + 20); p.post_b = F(b + 21); p.post_w = F(b + 22);
        p.pre_b = F(b + 23); p.pre_w = F(b + 24); p.q_b = F(b + 25); p.q_w = F(b + 26);
        p.qp_b = F(b + 27); p.qp_w = F(b + 28); p.t1_b = F(b + 29); p.t1_w = F(b + 30);
        p.t2_b = F(b + 31); p.t2_w = F(b + 32); p.t3_b = F(b + 33); p.t3_w = F(b + 34);
        p.tln_b = F(b + 35); p.tln_g = F(b + 36);
        p.edge = true;
    } else {
        const int b = 6 + 7 * 37;
        p.b_b = F(b + 0);  p.b_w = F(b + 1);  p.head_w = F(b + 2);
        p.kv_b = F(b + 3); p.kv_w = F(b + 4); p.kvp_b = F(b + 5); p.kvp_w = F(b + 6);
        p.ln_b = F(b + 7); p.ln_g = F(b + 8); p.out_b = F(b + 9); p.out_w = F(b + 10);
        p.post_b = F(b + 11); p.post_w = F(b + 12); p.pre_b = F(b + 13); p.pre_w = F(b + 14);
        p.q_b = F(b + 15); p.q_w = F(b + 16); p.qp_b = F(b + 17); p.qp_w = F(b + 18);
        p.t1_b = F(b + 19); p.t1_w = F(b + 20); p.t2_b = F(b + 21); p.t2_w = F(b + 22);
        p.t3_b = F(b + 23); p.t3_w = F(b + 24); p.tln_b = F(b + 25); p.tln_g = F(b + 26);
        p.edge = false;
    }
    return p;
}

struct LW {  // all transposed bf16 [N][K]
    __bf16 *q_w, *kv_w, *qp_w, *kvp_w, *b_w, *out_w, *pre_w, *t1_w, *t2_w, *t3_w, *post_w;
    __bf16 *e_init_w, *e1_w, *e2_w, *e_f_w;
};

static inline void launch_gemm(hipStream_t st, int AM, int BM,
    const void* A, int lda, const __bf16* ne,
    const __bf16* B, int ldb, const float* bias,
    float* Cf, __bf16* Cb, int ldc,
    const float* resid, int ldr,
    int M, int N, int K, float alpha, int act)
{
    dim3 grid((N + 63) / 64, (M + 127) / 128), block(256);
    if (AM == 0 && BM == 0)
        gemm_kernel<0, 0><<<grid, block, 0, st>>>(A, lda, ne, B, ldb, bias, Cf, Cb, ldc, resid, ldr, M, N, K, alpha, act);
    else if (AM == 0 && BM == 1)
        gemm_kernel<0, 1><<<grid, block, 0, st>>>(A, lda, ne, B, ldb, bias, Cf, Cb, ldc, resid, ldr, M, N, K, alpha, act);
    else if (AM == 1 && BM == 1)
        gemm_kernel<1, 1><<<grid, block, 0, st>>>(A, lda, ne, B, ldb, bias, Cf, Cb, ldc, resid, ldr, M, N, K, alpha, act);
    else
        gemm_kernel<3, 1><<<grid, block, 0, st>>>(A, lda, ne, B, ldb, bias, Cf, Cb, ldc, resid, ldr, M, N, K, alpha, act);
}

// workspace offsets (bytes, 256-aligned)
#define Z_OFF      (size_t)0            // f32  256*256*128
#define ZT_OFF     (size_t)33554432     // f32  256*256*128
#define H1_OFF     (size_t)67108864     // bf16 65536*256
#define H2_OFF     (size_t)100663296    // bf16 65536*256
#define ATTN_OFF   (size_t)134217728    // f32  4*256*256
#define BIASZ_OFF  (size_t)135266304    // f32  65536*4
#define FEATS_OFF  (size_t)136314880    // f32  256*768
#define S128_OFF   (size_t)137101312    // f32  256*128
#define S512A_OFF  (size_t)137232384    // f32  256*512
#define S512B_OFF  (size_t)137756672
#define S512C_OFF  (size_t)138280960
#define QBUFB_OFF  (size_t)138805248    // bf16 256*128
#define KVBUFB_OFF (size_t)138870784    // bf16 256*256
#define QPRAW_OFF  (size_t)139001856    // f32  256*48
#define KVPRAW_OFF (size_t)139051008    // f32  256*144
#define QPTS_OFF   (size_t)139198464    // f32  256*48
#define KVPTS_OFF  (size_t)139247616    // f32  256*144
#define KVPTSB_OFF (size_t)139395072    // bf16 256*144
#define NEB_OFF    (size_t)139468800    // bf16 256*64
#define OPTRAW_OFF (size_t)139501568    // f32  256*96
#define ZB_OFF     (size_t)139599872    // bf16 256*256*128
#define WB_OFF     (size_t)156377088    // bf16 weight pool (~19.9 MB)

extern "C" void kernel_launch(void* const* d_in, const int* in_sizes, int n_in,
                              void* d_out, int out_size, void* d_ws, size_t ws_size,
                              hipStream_t stream)
{
    (void)in_sizes; (void)n_in; (void)out_size; (void)ws_size;
    char* ws = (char*)d_ws;
    const float* rot   = (const float*)d_in[2];
    const float* trans = (const float*)d_in[3];
    const float* mask  = (const float*)d_in[4];

    float*  Z      = (float*)(ws + Z_OFF);
    float*  ZT     = (float*)(ws + ZT_OFF);
    __bf16* H1     = (__bf16*)(ws + H1_OFF);
    __bf16* H2     = (__bf16*)(ws + H2_OFF);
    float*  ATTN   = (float*)(ws + ATTN_OFF);
    float*  BIASZ  = (float*)(ws + BIASZ_OFF);
    float*  FEATS  = (float*)(ws + FEATS_OFF);
    float*  S128   = (float*)(ws + S128_OFF);
    float*  S512A  = (float*)(ws + S512A_OFF);
    float*  S512B  = (float*)(ws + S512B_OFF);
    float*  S512C  = (float*)(ws + S512C_OFF);
    __bf16* QBUFB  = (__bf16*)(ws + QBUFB_OFF);
    __bf16* KVBUFB = (__bf16*)(ws + KVBUFB_OFF);
    float*  QPRAW  = (float*)(ws + QPRAW_OFF);
    float*  KVPRAW = (float*)(ws + KVPRAW_OFF);
    float*  QPTS   = (float*)(ws + QPTS_OFF);
    float*  KVPTS  = (float*)(ws + KVPTS_OFF);
    __bf16* KVPTSB = (__bf16*)(ws + KVPTSB_OFF);
    __bf16* NEB    = (__bf16*)(ws + NEB_OFF);
    float*  OPTRAW = (float*)(ws + OPTRAW_OFF);
    __bf16* ZB     = (__bf16*)(ws + ZB_OFF);

    hipMemcpyAsync(S128, d_in[0], (size_t)NSEQ * CDIM * 4, hipMemcpyDeviceToDevice, stream);
    hipMemcpyAsync(Z,    d_in[1], (size_t)NSEQ * NSEQ * CDIM * 4, hipMemcpyDeviceToDevice, stream);
    cvt_kernel<<<32768, 256, 0, stream>>>((const float*)d_in[1], ZB, NSEQ * NSEQ * CDIM);

    // ---- pre-convert + pre-transpose all weights to bf16 [N][K] once ----
    LayerP lp[8];
    LW lw[8];
    {
        __bf16* wp = (__bf16*)(ws + WB_OFF);
        for (int li = 0; li < 8; ++li) {
            lp[li] = get_layer(d_in, li);
            auto CT = [&](const float* s, int K, int N) {
                __bf16* d = wp; wp += K * N;
                cvtT_kernel<<<(K * N + 255) / 256, 256, 0, stream>>>(s, d, K, N);
                return d;
            };
            lw[li].q_w    = CT(lp[li].q_w,    128, 128);
            lw[li].kv_w   = CT(lp[li].kv_w,   128, 256);
            lw[li].qp_w   = CT(lp[li].qp_w,   128, 48);
            lw[li].kvp_w  = CT(lp[li].kvp_w,  128, 144);
            lw[li].b_w    = CT(lp[li].b_w,    128, 4);
            lw[li].out_w  = CT(lp[li].out_w,  768, 128);
            lw[li].pre_w  = CT(lp[li].pre_w,  128, 512);
            lw[li].t1_w   = CT(lp[li].t1_w,   512, 512);
            lw[li].t2_w   = CT(lp[li].t2_w,   512, 512);
            lw[li].t3_w   = CT(lp[li].t3_w,   512, 512);
            lw[li].post_w = CT(lp[li].post_w, 512, 128);
            if (lp[li].edge) {
                lw[li].e_init_w = CT(lp[li].e_init_w, 128, 64);
                lw[li].e1_w     = CT(lp[li].e1_w,     256, 256);
                lw[li].e2_w     = CT(lp[li].e2_w,     256, 256);
                lw[li].e_f_w    = CT(lp[li].e_f_w,    256, 128);
            }
        }
    }

    const float qk_scale = 0.10206207261596575f;  // sqrt(1/96)

    for (int li = 0; li < 8; ++li) {
        const LayerP& p = lp[li];
        const LW& w = lw[li];

        // ---- IPA projections (q/kv in bf16: only consumed as WMMA operands) ----
        launch_gemm(stream, 0, 1, S128, 128, nullptr, w.q_w,  128, p.q_b,  nullptr, QBUFB,  128, nullptr, 0, 256, 128, 128, 1.f, 0);
        launch_gemm(stream, 0, 1, S128, 128, nullptr, w.kv_w, 128, p.kv_b, nullptr, KVBUFB, 256, nullptr, 0, 256, 256, 128, 1.f, 0);
        launch_gemm(stream, 0, 1, S128, 128, nullptr, w.qp_w, 128, p.qp_b,  QPRAW,  nullptr,  48, nullptr, 0, 256,  48, 128, 1.f, 0);
        launch_gemm(stream, 0, 1, S128, 128, nullptr, w.kvp_w, 128, p.kvp_b, KVPRAW, nullptr, 144, nullptr, 0, 256, 144, 128, 1.f, 0);
        pts_kernel<<<256, 64, 0, stream>>>(QPRAW, KVPRAW, rot, trans, QPTS, KVPTS, KVPTSB);

        // ---- pair bias: z @ b_w (bf16 z shadow) ----
        launch_gemm(stream, 1, 1, ZB, 128, nullptr, w.b_w, 128, p.b_b, BIASZ, nullptr, 4, nullptr, 0, 65536, 4, 128, 1.f, 0);

        // ---- logits per head: q_h @ k_h^T ----
        for (int h = 0; h < 4; ++h)
            launch_gemm(stream, 1, 1, QBUFB + h * 32, 128, nullptr, KVBUFB + h * 64, 256, nullptr,
                        ATTN + (size_t)h * 65536, nullptr, 256, nullptr, 0, 256, 256, 32, qk_scale, 0);

        softmax_kernel<<<dim3(256, 4), 256, 0, stream>>>(ATTN, BIASZ, QPTS, KVPTS, p.head_w, mask);

        // ---- o = a @ v  and  o_pt = a @ v_pts (activation B: BMODE 0) ----
        for (int h = 0; h < 4; ++h)
            launch_gemm(stream, 0, 0, ATTN + (size_t)h * 65536, 256, nullptr, KVBUFB + h * 64 + 32, 256, nullptr,
                        FEATS + h * 32, nullptr, 768, nullptr, 0, 256, 32, 256, 1.f, 0);
        for (int h = 0; h < 4; ++h)
            launch_gemm(stream, 0, 0, ATTN + (size_t)h * 65536, 256, nullptr, KVPTSB + (h * 12 + 4) * 3, 144, nullptr,
                        OPTRAW + h * 24, nullptr, 96, nullptr, 0, 256, 24, 256, 1.f, 0);

        opair_kernel<<<256, 128, 0, stream>>>(ATTN, Z, FEATS);
        optfin_kernel<<<256, 32, 0, stream>>>(OPTRAW, rot, trans, FEATS);

        // ---- IPA out + residual, layernorm ----
        launch_gemm(stream, 0, 1, FEATS, 768, nullptr, w.out_w, 768, p.out_b, S128, nullptr, 128, S128, 128, 256, 128, 768, 1.f, 0);
        ln_kernel<<<256, 128, 0, stream>>>(S128, S128, nullptr, p.ln_g, p.ln_b, 128);

        // ---- transition ----
        launch_gemm(stream, 0, 1, S128,  128, nullptr, w.pre_w, 128, p.pre_b, S512A, nullptr, 512, nullptr, 0, 256, 512, 128, 1.f, 0);
        launch_gemm(stream, 0, 1, S512A, 512, nullptr, w.t1_w,  512, p.t1_b,  S512B, nullptr, 512, nullptr, 0, 256, 512, 512, 1.f, 1);
        launch_gemm(stream, 0, 1, S512B, 512, nullptr, w.t2_w,  512, p.t2_b,  S512C, nullptr, 512, nullptr, 0, 256, 512, 512, 1.f, 1);
        launch_gemm(stream, 0, 1, S512C, 512, nullptr, w.t3_w,  512, p.t3_b,  S512A, nullptr, 512, S512A, 512, 256, 512, 512, 1.f, 0);
        ln_kernel<<<256, 128, 0, stream>>>(S512A, S512A, nullptr, p.tln_g, p.tln_b, 512);
        launch_gemm(stream, 0, 1, S512A, 512, nullptr, w.post_w, 512, p.post_b, S128, nullptr, 128, nullptr, 0, 256, 128, 512, 1.f, 0);

        // ---- edge transition (bf16 end-to-end through the WMMA path) ----
        if (p.edge) {
            launch_gemm(stream, 0, 1, S128, 128, nullptr, w.e_init_w, 128, p.e_init_b, nullptr, NEB, 64, nullptr, 0, 256, 64, 128, 1.f, 0);
            launch_gemm(stream, 3, 1, ZB, 128, NEB, w.e1_w, 256, p.e1_b, nullptr, H1, 256, nullptr, 0, 65536, 256, 256, 1.f, 1);
            launch_gemm(stream, 1, 1, H1, 256, nullptr, w.e2_w, 256, p.e2_b, nullptr, H2, 256, nullptr, 0, 65536, 256, 256, 1.f, 1);
            launch_gemm(stream, 1, 1, H2, 256, nullptr, w.e_f_w, 256, p.e_f_b, ZT, nullptr, 128, nullptr, 0, 65536, 128, 256, 1.f, 0);
            ln_kernel<<<65536, 128, 0, stream>>>(ZT, Z, ZB, p.e_ln_g, p.e_ln_b, 128);
        }
    }

    hipMemcpyAsync(d_out, S128, (size_t)NSEQ * CDIM * 4, hipMemcpyDeviceToDevice, stream);
}
